// TemporalGATGRU_21792664060094
// MI455X (gfx1250) — compile-verified, matrix operand order checked by reading, not run on hardware
//
#include <hip/hip_runtime.h>
#include <math.h>

#define N_NODES 100000
#define E_EDGES 800000
#define ET_EDGES (E_EDGES + N_NODES)
#define HID 64
#define F_IN 128
#define HEADS 4

typedef __bf16 bf16_t;
typedef __attribute__((ext_vector_type(16))) __bf16 v16bf;
typedef __attribute__((ext_vector_type(8)))  __bf16 v8bf;
typedef __attribute__((ext_vector_type(8)))  float  v8f;

__device__ __forceinline__ float lrelu02(float v) { return v > 0.f ? v : 0.2f * v; }
__device__ __forceinline__ float sigm(float x) { return 1.f / (1.f + expf(-x)); }

__device__ __forceinline__ void atomicMaxFloat(float* addr, float val) {
    // works with buffer initialized to -inf (0xFF800000)
    if (val >= 0.f) atomicMax((int*)addr, __float_as_int(val));
    else            atomicMin((unsigned int*)addr, __float_as_uint(val));
}

// ---------------- fill ----------------
__global__ void k_fill_f32(float* p, float v, long long n) {
    long long i = blockIdx.x * (long long)blockDim.x + threadIdx.x;
    if (i < n) p[i] = v;
}

// ---------------- f32 -> bf16 ----------------
__global__ void k_f32_to_bf16(const float* __restrict__ src, bf16_t* __restrict__ dst, long long n) {
    long long i = blockIdx.x * (long long)blockDim.x + threadIdx.x;
    if (i < n) dst[i] = (bf16_t)src[i];
}

// ---------------- pack weight matrix into WMMA B-operand layout ----------------
// Packed layout: Bp[(((kt*CT + ct)*32 + lane)*16) + i] = B[kt*32 + kmap(i,lane>>4)][ct*16 + (lane&15)]
__global__ void k_pack_b(const float* __restrict__ Bsrc, bf16_t* __restrict__ Bp,
                         int K, int NC, int transpose) {
    int flat = blockIdx.x * blockDim.x + threadIdx.x;
    int total = K * NC;
    if (flat >= total) return;
    int CT   = NC >> 4;
    int i    = flat & 15;
    int lane = (flat >> 4) & 31;
    int tc   = flat >> 9;          // 16*32 = 512 elems per (kt,ct) tile
    int ct   = tc % CT;
    int kt   = tc / CT;
    int lh   = lane >> 4;
    int n    = ct * 16 + (lane & 15);
    int k    = kt * 32 + ((i < 8) ? (8 * lh + i) : (16 + 8 * lh + (i - 8)));
    float v  = transpose ? Bsrc[n * K + k] : Bsrc[k * NC + n];
    Bp[flat] = (bf16_t)v;
}

// ---------------- bf16 WMMA GEMM: C[M,NC] = A[M,K] @ B[K,NC] (+bias, +relu) ----------------
// 128-thread blocks = 4 waves; each wave owns one 16-row tile and CTW column tiles.
// The block's entire B panel is staged once into LDS (shared by all 4 waves), so the
// WMMA inner loop feeds its B operands from short-latency ds_load_b128 instead of L2.
template <int CTW>
__global__ __launch_bounds__(128) void k_gemm_bf16(const bf16_t* __restrict__ A,
                                                   const bf16_t* __restrict__ Bp,
                                                   const float* __restrict__ bias,
                                                   float* __restrict__ C,
                                                   int M, int K, int NC, int act) {
    extern __shared__ bf16_t Bs[];  // kTiles * CTW * 512 bf16
    const int CT     = NC >> 4;
    const int kTiles = K >> 5;
    const int ct0    = blockIdx.y * CTW;
    const int chunk  = CTW * 512;   // elems per k-step panel slice

    // cooperative copy: B panel (this block's ct0..ct0+CTW-1 tiles, all k-steps) -> LDS
    for (int kt = 0; kt < kTiles; ++kt) {
        const bf16_t* srcp = Bp + ((size_t)(kt * CT + ct0) << 9);
        bf16_t* dstp = Bs + kt * chunk;
        for (int idx = threadIdx.x * 8; idx < chunk; idx += 128 * 8)
            *(v8bf*)(dstp + idx) = *(const v8bf*)(srcp + idx);
    }
    __syncthreads();

    const int lane = threadIdx.x & 31;  // lane within wave
    const int wv   = threadIdx.x >> 5;  // wave index in block (0..3)
    const int lh   = lane >> 4;
    const int ml   = lane & 15;
    const int rt   = blockIdx.x * 4 + wv;
    if (rt * 16 >= M) return;           // wave-uniform; after the barrier

    const int m = rt * 16 + ml;
    const bf16_t* Arow = A + (size_t)m * K;

    v8f acc[CTW];
#pragma unroll
    for (int i = 0; i < CTW; ++i) {
#pragma unroll
        for (int j = 0; j < 8; ++j) acc[i][j] = 0.f;
    }

    for (int kt = 0; kt < kTiles; ++kt) {
        const int kb = kt << 5;
        v8bf alo = *(const v8bf*)(Arow + kb + 8 * lh);
        v8bf ahi = *(const v8bf*)(Arow + kb + 16 + 8 * lh);
        v16bf afrag = __builtin_shufflevector(alo, ahi, 0, 1, 2, 3, 4, 5, 6, 7,
                                              8, 9, 10, 11, 12, 13, 14, 15);
        const bf16_t* bbase = Bs + kt * chunk + lane * 16;
#pragma unroll
        for (int ct = 0; ct < CTW; ++ct) {
            v16bf bfrag = *(const v16bf*)(bbase + ct * 512);  // ds_load from LDS
            acc[ct] = __builtin_amdgcn_wmma_f32_16x16x32_bf16(
                false, afrag, false, bfrag, (short)0, acc[ct], false, false);
        }
    }

    // D layout: VGPR r -> row = rt*16 + 8*lh + r, col = ct*16 + (lane&15)
    const int rowBase = rt * 16 + lh * 8;
#pragma unroll
    for (int ct = 0; ct < CTW; ++ct) {
        const int col = (ct0 + ct) * 16 + ml;
        const float bv = bias ? bias[col] : 0.f;
#pragma unroll
        for (int r = 0; r < 8; ++r) {
            float v = acc[ct][r] + bv;
            if (act) v = fmaxf(v, 0.f);
            C[(size_t)(rowBase + r) * NC + col] = v;
        }
    }
}

// ---------------- GAT attention scores: a_src[n,h], a_dst[n,h] ----------------
__global__ void k_gat_scores(const float* __restrict__ h, const float* __restrict__ att_s,
                             const float* __restrict__ att_d, float* __restrict__ a_s,
                             float* __restrict__ a_d, int nNodes, int H) {
    int t = blockIdx.x * blockDim.x + threadIdx.x;
    if (t >= nNodes * H) return;
    int n = t / H, hd = t % H;
    const float* hp = h + (size_t)n * H * HID + (size_t)hd * HID;
    const float* ws = att_s + hd * HID;
    const float* wd = att_d + hd * HID;
    float ss = 0.f, sd = 0.f;
    for (int c = 0; c < HID; ++c) { float v = hp[c]; ss += v * ws[c]; sd += v * wd[c]; }
    a_s[t] = ss;
    a_d[t] = sd;
}

// ---------------- edge pass 1: segment max over dst ----------------
__global__ void k_edge_max(const int* __restrict__ src, const int* __restrict__ dst,
                           const float* __restrict__ a_s, const float* __restrict__ a_d,
                           float* mbuf, int H) {
    int e = blockIdx.x * blockDim.x + threadIdx.x;
    if (e >= ET_EDGES) return;
    int s, d;
    if (e < E_EDGES) { s = src[e]; d = dst[e]; } else { s = d = e - E_EDGES; }
    for (int hd = 0; hd < H; ++hd) {
        float v = lrelu02(a_s[(size_t)s * H + hd] + a_d[(size_t)d * H + hd]);
        atomicMaxFloat(&mbuf[(size_t)d * H + hd], v);
    }
}

// ---------------- edge pass 2: p = exp(e - max), segment sum ----------------
__global__ void k_edge_p(const int* __restrict__ src, const int* __restrict__ dst,
                         const float* __restrict__ a_s, const float* __restrict__ a_d,
                         const float* __restrict__ mbuf, float* __restrict__ pbuf,
                         float* denom, int H) {
    int e = blockIdx.x * blockDim.x + threadIdx.x;
    if (e >= ET_EDGES) return;
    int s, d;
    if (e < E_EDGES) { s = src[e]; d = dst[e]; } else { s = d = e - E_EDGES; }
    for (int hd = 0; hd < H; ++hd) {
        float v = lrelu02(a_s[(size_t)s * H + hd] + a_d[(size_t)d * H + hd]);
        float p = expf(v - mbuf[(size_t)d * H + hd]);
        pbuf[(size_t)e * H + hd] = p;
        atomicAdd(&denom[(size_t)d * H + hd], p);
    }
}

// ---------------- edge pass 2.5: alpha = p / (denom + eps), in place ----------------
__global__ void k_edge_alpha(const int* __restrict__ dst, const float* __restrict__ denom,
                             float* pbuf, int H) {
    int t = blockIdx.x * blockDim.x + threadIdx.x;
    if (t >= ET_EDGES * H) return;
    int e = t / H, hd = t % H;
    int d = (e < E_EDGES) ? dst[e] : (e - E_EDGES);
    pbuf[t] = pbuf[t] / (denom[(size_t)d * H + hd] + 1e-16f);
}

// ---------------- edge pass 3: out[d] += h[s] * alpha ----------------
__global__ void k_edge_scatter(const int* __restrict__ src, const int* __restrict__ dst,
                               const float* __restrict__ h, const float* __restrict__ alpha,
                               float* out, int H) {
    const int HC = H * HID;
    long long t = blockIdx.x * (long long)blockDim.x + threadIdx.x;
    long long total = (long long)ET_EDGES * HC;
    if (t >= total) return;
    int e = (int)(t / HC);
    int j = (int)(t % HC);
    int hd = j / HID;
    int s, d;
    if (e < E_EDGES) { s = src[e]; d = dst[e]; } else { s = d = e - E_EDGES; }
    float a = alpha[(size_t)e * H + hd];
    atomicAdd(&out[(size_t)d * HC + j], h[(size_t)s * HC + j] * a);
}

// ---------------- bias (+relu) + convert to bf16 ----------------
__global__ void k_bias_act_bf16(const float* __restrict__ acc, const float* __restrict__ bias,
                                bf16_t* __restrict__ outb, long long total, int C, int relu) {
    long long t = blockIdx.x * (long long)blockDim.x + threadIdx.x;
    if (t >= total) return;
    int c = (int)(t % C);
    float v = acc[t] + bias[c];
    if (relu) v = fmaxf(v, 0.f);
    outb[t] = (bf16_t)v;
}

// ---------------- GRU elementwise (h_prev = 0) ----------------
__global__ void k_gru_eltwise(const float* __restrict__ gi, const float* __restrict__ bhh,
                              float* hf, bf16_t* hb, int nNodes) {
    int t = blockIdx.x * blockDim.x + threadIdx.x;
    if (t >= nNodes * HID) return;
    int n = t / HID, c = t % HID;
    const float* g = gi + (size_t)n * 3 * HID;
    float r  = sigm(g[c] + bhh[c]);
    float z  = sigm(g[HID + c] + bhh[HID + c]);
    float nn = tanhf(g[2 * HID + c] + r * bhh[2 * HID + c]);
    float hv = (1.f - z) * nn;  // + z*h0 with h0 == 0
    if (hf) hf[t] = hv;
    if (hb) hb[t] = (bf16_t)hv;
}

// ---------------- final fc: out[n,3] = h[n,:] @ fcW^T + fcb ----------------
__global__ void k_fc(const float* __restrict__ h, const float* __restrict__ W,
                     const float* __restrict__ b, float* __restrict__ out, int nNodes) {
    int n = blockIdx.x * blockDim.x + threadIdx.x;
    if (n >= nNodes) return;
    const float* hp = h + (size_t)n * HID;
    for (int j = 0; j < 3; ++j) {
        float s = b[j];
        const float* w = W + j * HID;
        for (int c = 0; c < HID; ++c) s += hp[c] * w[c];
        out[(size_t)n * 3 + j] = s;
    }
}

static inline dim3 grid1(long long n, int bs = 256) { return dim3((unsigned)((n + bs - 1) / bs)); }

extern "C" void kernel_launch(void* const* d_in, const int* in_sizes, int n_in,
                              void* d_out, int out_size, void* d_ws, size_t ws_size,
                              hipStream_t stream) {
    (void)in_sizes; (void)n_in; (void)out_size; (void)ws_size;
    const float* x      = (const float*)d_in[0];
    const int*   ei     = (const int*)d_in[1];
    const int*   src    = ei;
    const int*   dst    = ei + E_EDGES;
    const float* W1     = (const float*)d_in[2];
    const float* attS1  = (const float*)d_in[3];
    const float* attD1  = (const float*)d_in[4];
    const float* b1     = (const float*)d_in[5];
    const float* W2     = (const float*)d_in[6];
    const float* attS2  = (const float*)d_in[7];
    const float* attD2  = (const float*)d_in[8];
    const float* b2     = (const float*)d_in[9];
    const float* Wih0   = (const float*)d_in[10];
    const float* bih0   = (const float*)d_in[12];
    const float* bhh0   = (const float*)d_in[13];
    const float* Wih1   = (const float*)d_in[14];
    const float* bih1   = (const float*)d_in[16];
    const float* bhh1   = (const float*)d_in[17];
    const float* fcW    = (const float*)d_in[18];
    const float* fcb    = (const float*)d_in[19];
    float* out = (float*)d_out;

    // ---- workspace bump allocator with phase-ordered reuse ----
    char* ws = (char*)d_ws;
    size_t off = 0;
    auto take = [&](size_t bytes) -> char* {
        char* p = ws + off;
        off += (bytes + 255) & ~(size_t)255;
        return p;
    };
    float*  R1   = (float*)take((size_t)N_NODES * 256 * 4);  // h1 -> h2/out2 -> hfinal
    float*  R2   = (float*)take((size_t)N_NODES * 256 * 4);  // out1 -> gi -> gi2
    bf16_t* R3   = (bf16_t*)take((size_t)N_NODES * 256 * 2); // xb -> x1b -> x2b -> hb
    float*  aS   = (float*)take((size_t)N_NODES * HEADS * 4);
    float*  aD   = (float*)take((size_t)N_NODES * HEADS * 4);
    float*  mbuf = (float*)take((size_t)N_NODES * HEADS * 4);
    float*  den  = (float*)take((size_t)N_NODES * HEADS * 4);
    float*  pbuf = (float*)take((size_t)ET_EDGES * HEADS * 4);
    bf16_t* W1p  = (bf16_t*)take((size_t)F_IN * 256 * 2);
    bf16_t* W2p  = (bf16_t*)take((size_t)256 * HID * 2);
    bf16_t* Wi0p = (bf16_t*)take((size_t)HID * 192 * 2);
    bf16_t* Wi1p = (bf16_t*)take((size_t)HID * 192 * 2);

    const int MT = N_NODES / 16;        // 6250 row tiles (exact)
    const int GEMM_BLKS = (MT + 3) / 4; // 4 row-tile waves per 128-thread block

    // dynamic LDS sizes: kTiles * CTW * 512 bf16 * 2 bytes
    const size_t SM1 = (size_t)(F_IN / 32) * 8 * 512 * 2;  // 32 KB (K=128, CTW=8)
    const size_t SM2 = (size_t)(256 / 32) * 4 * 512 * 2;   // 32 KB (K=256, CTW=4)
    const size_t SMG = (size_t)(HID / 32) * 6 * 512 * 2;   // 12 KB (K=64,  CTW=6)

    // ---- prep: convert x, pack all weights ----
    k_f32_to_bf16<<<grid1((long long)N_NODES * F_IN), 256, 0, stream>>>(x, R3, (long long)N_NODES * F_IN);
    k_pack_b<<<grid1(F_IN * 256), 256, 0, stream>>>(W1, W1p, F_IN, 256, 0);
    k_pack_b<<<grid1(256 * HID), 256, 0, stream>>>(W2, W2p, 256, HID, 0);
    k_pack_b<<<grid1(HID * 192), 256, 0, stream>>>(Wih0, Wi0p, HID, 192, 1);  // B = Wih0^T
    k_pack_b<<<grid1(HID * 192), 256, 0, stream>>>(Wih1, Wi1p, HID, 192, 1);

    // ---- GAT layer 1 ----
    k_gemm_bf16<8><<<dim3(GEMM_BLKS, 2), 128, SM1, stream>>>(R3, W1p, nullptr, R1, N_NODES, F_IN, 256, 0);
    k_gat_scores<<<grid1((long long)N_NODES * HEADS), 256, 0, stream>>>(R1, attS1, attD1, aS, aD, N_NODES, HEADS);
    k_fill_f32<<<grid1((long long)N_NODES * HEADS), 256, 0, stream>>>(mbuf, -INFINITY, (long long)N_NODES * HEADS);
    k_fill_f32<<<grid1((long long)N_NODES * HEADS), 256, 0, stream>>>(den, 0.f, (long long)N_NODES * HEADS);
    k_fill_f32<<<grid1((long long)N_NODES * 256), 256, 0, stream>>>(R2, 0.f, (long long)N_NODES * 256);
    k_edge_max<<<grid1(ET_EDGES), 256, 0, stream>>>(src, dst, aS, aD, mbuf, HEADS);
    k_edge_p<<<grid1(ET_EDGES), 256, 0, stream>>>(src, dst, aS, aD, mbuf, pbuf, den, HEADS);
    k_edge_alpha<<<grid1((long long)ET_EDGES * HEADS), 256, 0, stream>>>(dst, den, pbuf, HEADS);
    k_edge_scatter<<<grid1((long long)ET_EDGES * HEADS * HID), 256, 0, stream>>>(src, dst, R1, pbuf, R2, HEADS);
    // x1 = relu(out1 + b1) -> bf16
    k_bias_act_bf16<<<grid1((long long)N_NODES * 256), 256, 0, stream>>>(R2, b1, R3, (long long)N_NODES * 256, 256, 1);

    // ---- GAT layer 2 (1 head) ----
    float* h2   = R1;                          // N x 64
    float* out2 = R1 + (size_t)N_NODES * HID;  // N x 64
    k_gemm_bf16<4><<<dim3(GEMM_BLKS, 1), 128, SM2, stream>>>(R3, W2p, nullptr, h2, N_NODES, 256, HID, 0);
    k_gat_scores<<<grid1(N_NODES), 256, 0, stream>>>(h2, attS2, attD2, aS, aD, N_NODES, 1);
    k_fill_f32<<<grid1(N_NODES), 256, 0, stream>>>(mbuf, -INFINITY, N_NODES);
    k_fill_f32<<<grid1(N_NODES), 256, 0, stream>>>(den, 0.f, N_NODES);
    k_fill_f32<<<grid1((long long)N_NODES * HID), 256, 0, stream>>>(out2, 0.f, (long long)N_NODES * HID);
    k_edge_max<<<grid1(ET_EDGES), 256, 0, stream>>>(src, dst, aS, aD, mbuf, 1);
    k_edge_p<<<grid1(ET_EDGES), 256, 0, stream>>>(src, dst, aS, aD, mbuf, pbuf, den, 1);
    k_edge_alpha<<<grid1(ET_EDGES), 256, 0, stream>>>(dst, den, pbuf, 1);
    k_edge_scatter<<<grid1((long long)ET_EDGES * HID), 256, 0, stream>>>(src, dst, h2, pbuf, out2, 1);
    // x2 = out2 + b2 -> bf16
    k_bias_act_bf16<<<grid1((long long)N_NODES * HID), 256, 0, stream>>>(out2, b2, R3, (long long)N_NODES * HID, HID, 0);

    // ---- GRU layer 0 (h0 = 0): gi = x2 @ Wih0^T + bih0 ----
    k_gemm_bf16<6><<<dim3(GEMM_BLKS, 2), 128, SMG, stream>>>(R3, Wi0p, bih0, R2, N_NODES, HID, 192, 0);
    k_gru_eltwise<<<grid1((long long)N_NODES * HID), 256, 0, stream>>>(R2, bhh0, nullptr, R3, N_NODES);

    // ---- GRU layer 1 ----
    k_gemm_bf16<6><<<dim3(GEMM_BLKS, 2), 128, SMG, stream>>>(R3, Wi1p, bih1, R2, N_NODES, HID, 192, 0);
    float* hfinal = R1;
    k_gru_eltwise<<<grid1((long long)N_NODES * HID), 256, 0, stream>>>(R2, bhh1, hfinal, nullptr, N_NODES);

    // ---- fc ----
    k_fc<<<grid1(N_NODES), 256, 0, stream>>>(hfinal, fcW, fcb, out, N_NODES);
}
